// ProteinFeatures_2353642078944
// MI455X (gfx1250) — matrix operand order repaired and checked by reading, not compiled, not run on hardware
//
#include <hip/hip_runtime.h>
#include <hip/hip_bf16.h>

typedef __attribute__((ext_vector_type(16))) _Float16 v16h;
typedef __attribute__((ext_vector_type(8)))  _Float16 v8h;
typedef __attribute__((ext_vector_type(8)))  float    v8f;

#define LSEQ   2048
#define TOPK   30
#define KDIM   416
#define NDIM   128
#define NPAIR  24
#define RPD    2                 // residues per block in feature kernel
#define ROWS   (RPD*TOPK)        // 60 edges per block
#define ROWS_PAD 64

// atom indices: 0=N, 1=C, 2=Ca, 3=Cb, 4=O
__constant__ int c_pa[NPAIR] = {0,2,3,1,1,1,0,0,3,0,2,3,2,3,2,4,1,4,0,4,3,4,4,2};
__constant__ int c_pb[NPAIR] = {0,2,3,0,2,3,2,3,2,1,1,1,0,0,3,4,4,1,4,0,4,2,3,4};

__device__ inline void make_atoms(const float* __restrict__ Xr, float dst[5][3]) {
    // Xr points at X[gi*12]: N(0..2) C(3..5) Ca(6..8) O(9..11)
    float bx = Xr[6]-Xr[0], by = Xr[7]-Xr[1], bz = Xr[8]-Xr[2];   // Ca - N
    float cx = Xr[3]-Xr[6], cy = Xr[4]-Xr[7], cz = Xr[5]-Xr[8];   // C - Ca
    float ax = by*cz - bz*cy;
    float ay = bz*cx - bx*cz;
    float az = bx*cy - by*cx;
    dst[0][0]=Xr[0]; dst[0][1]=Xr[1];  dst[0][2]=Xr[2];
    dst[1][0]=Xr[3]; dst[1][1]=Xr[4];  dst[1][2]=Xr[5];
    dst[2][0]=Xr[6]; dst[2][1]=Xr[7];  dst[2][2]=Xr[8];
    dst[3][0]=-0.58273431f*ax + 0.56802827f*bx - 0.54067466f*cx + Xr[6];
    dst[3][1]=-0.58273431f*ay + 0.56802827f*by - 0.54067466f*cy + Xr[7];
    dst[3][2]=-0.58273431f*az + 0.56802827f*bz - 0.54067466f*cz + Xr[8];
    dst[4][0]=Xr[9]; dst[4][1]=Xr[10]; dst[4][2]=Xr[11];
}

// ---------------------------------------------------------------------------
// Kernel 0: edge_w (416x128 f32, K-major) -> Wt (128x416 f16, N-major rows)
// ---------------------------------------------------------------------------
__global__ __launch_bounds__(256) void prep_w_kernel(const float* __restrict__ ew,
                                                     _Float16* __restrict__ Wt) {
    int idx = blockIdx.x * 256 + threadIdx.x;
    if (idx < KDIM * NDIM) {
        int n  = idx / KDIM;
        int kd = idx % KDIM;
        Wt[idx] = (_Float16)ew[kd * NDIM + n];
    }
}

// ---------------------------------------------------------------------------
// Kernel 1: masked distances + top-k (ascending, lowest index wins ties)
// one 256-thread block per (b,i) row
// ---------------------------------------------------------------------------
__global__ __launch_bounds__(256) void topk_kernel(
    const float* __restrict__ X, const float* __restrict__ mask,
    int* __restrict__ eidx, float* __restrict__ dnb, float* __restrict__ out_eidx)
{
    __shared__ float              s_dist[LSEQ];
    __shared__ unsigned long long s_key[LSEQ];
    __shared__ unsigned long long s_redu[256];
    __shared__ float              s_redf[256];

    const int tid = threadIdx.x;
    const int gi  = blockIdx.x;          // b*L + i
    const int b   = gi / LSEQ;
    const float mi = mask[gi];
    const float cx = X[(gi*4+1)*3+0];    // C atom of residue i
    const float cy = X[(gi*4+1)*3+1];
    const float cz = X[(gi*4+1)*3+2];

    float lmax = 0.0f;
    for (int j = tid; j < LSEQ; j += 256) {
        int gj = b*LSEQ + j;
        float dx = X[(gj*4+1)*3+0]-cx;
        float dy = X[(gj*4+1)*3+1]-cy;
        float dz = X[(gj*4+1)*3+2]-cz;
        float d  = sqrtf(dx*dx + dy*dy + dz*dz + 1e-6f);
        float Dj = mi * mask[gj] * d;
        s_dist[j] = Dj;
        lmax = fmaxf(lmax, Dj);
    }
    s_redf[tid] = lmax; __syncthreads();
    for (int s = 128; s > 0; s >>= 1) {
        if (tid < s) s_redf[tid] = fmaxf(s_redf[tid], s_redf[tid+s]);
        __syncthreads();
    }
    const float dmax = s_redf[0];

    for (int j = tid; j < LSEQ; j += 256) {
        float m2   = mi * mask[b*LSEQ + j];
        float Dadj = s_dist[j] + (1.0f - m2) * dmax;   // >= 0 -> bits monotonic
        s_key[j] = ((unsigned long long)__float_as_uint(Dadj) << 32) | (unsigned int)j;
    }
    __syncthreads();

    for (int r = 0; r < TOPK; r++) {
        unsigned long long lmin = ~0ULL;
        #pragma unroll
        for (int q = 0; q < 8; q++) {
            unsigned long long kk = s_key[tid*8 + q];
            lmin = (kk < lmin) ? kk : lmin;
        }
        s_redu[tid] = lmin; __syncthreads();
        for (int s = 128; s > 0; s >>= 1) {
            if (tid < s) {
                unsigned long long a = s_redu[tid], c = s_redu[tid+s];
                s_redu[tid] = (c < a) ? c : a;
            }
            __syncthreads();
        }
        if (tid == 0) {
            unsigned long long w = s_redu[0];
            unsigned int j = (unsigned int)(w & 0xffffffffu);
            int off = gi*TOPK + r;
            eidx[off]     = (int)j;
            dnb[off]      = __uint_as_float((unsigned int)(w >> 32));
            out_eidx[off] = (float)j;
            s_key[j] = ~0ULL;
        }
        __syncthreads();
    }
}

// ---------------------------------------------------------------------------
// Kernel 2: features (f16 in LDS) -> WMMA GEMM (64x416)x(416x128) -> LayerNorm
// 256 threads = 8 wave32; wave w: M-tile (w&3), N-half (w>>2)
// ---------------------------------------------------------------------------
__global__ __launch_bounds__(256) void feat_kernel(
    const float* __restrict__ X,
    const int*   __restrict__ residue_idx,
    const int*   __restrict__ chain_labels,
    const float* __restrict__ pe_w, const float* __restrict__ pe_b,
    const float* __restrict__ ln_g, const float* __restrict__ ln_b,
    const int*   __restrict__ eidx, const float* __restrict__ dnb,
    const _Float16* __restrict__ Wt,
    float* __restrict__ out)
{
    __shared__ __align__(32) _Float16 sF[ROWS_PAD * KDIM];  // 53248 B; reused as Cbuf
    __shared__ float s_nb[ROWS][5][3];
    __shared__ float s_ct[RPD][5][3];
    __shared__ float s_dn[ROWS];
    __shared__ int   s_di[ROWS];
    __shared__ float s_mu[ROWS_PAD];
    __shared__ float s_rs[ROWS_PAD];

    const int tid = threadIdx.x;
    const int gi0 = blockIdx.x * RPD;

    // Warm L2/near caches for the weight panel (global_prefetch_b8); each
    // thread covers a 416 B slice -> whole 106 KB Wt panel per block.
    __builtin_prefetch(((const char*)Wt) + tid * 416, 0, 3);

    // ---- stage neighbor atoms + per-edge scalars ----
    if (tid < ROWS) {
        int e   = tid;
        int gi  = gi0 + e / TOPK;
        int b   = gi / LSEQ;
        int kk  = e % TOPK;
        int off = gi*TOPK + kk;
        int j   = eidx[off];
        int gj  = b*LSEQ + j;
        make_atoms(&X[(size_t)gj*12], s_nb[e]);
        s_dn[e] = dnb[off];
        int offr = residue_idx[gi] - residue_idx[gj];
        int ec   = (chain_labels[gi] == chain_labels[gj]) ? 1 : 0;
        int d    = offr + 32; d = d < 0 ? 0 : (d > 64 ? 64 : d);
        s_di[e]  = ec ? d : 65;
    } else if (tid >= 64 && tid < 64 + RPD) {
        int r  = tid - 64;
        make_atoms(&X[(size_t)(gi0 + r)*12], s_ct[r]);
    }
    __syncthreads();

    // ---- fill 64x416 f16 feature matrix: 26 groups of 16 cols per row ----
    for (int task = tid; task < ROWS_PAD * 26; task += 256) {
        int row = task / 26;
        int g   = task % 26;
        float vals[16];
        if (row >= ROWS) {
            #pragma unroll
            for (int c = 0; c < 16; c++) vals[c] = 0.0f;
        } else if (g == 0) {                       // positional embedding
            int d = s_di[row];
            #pragma unroll
            for (int c = 0; c < 16; c++) vals[c] = pe_w[d*16 + c] + pe_b[c];
        } else {                                   // RBF groups
            float dd;
            if (g == 1) {
                dd = s_dn[row];
            } else {
                int p = g - 2;
                const float* A  = s_ct[row / TOPK][c_pa[p]];
                const float* Bp = s_nb[row][c_pb[p]];
                float dx = A[0]-Bp[0], dy = A[1]-Bp[1], dz = A[2]-Bp[2];
                dd = sqrtf(dx*dx + dy*dy + dz*dz + 1e-6f);
            }
            #pragma unroll
            for (int c = 0; c < 16; c++) {
                float t = (dd - (2.0f + (20.0f/15.0f)*c)) * 0.8f;  // mu=linspace(2,22,16), sigma=1.25
                vals[c] = __expf(-t*t);
            }
        }
        v8h h0, h1;
        #pragma unroll
        for (int c = 0; c < 8; c++) { h0[c] = (_Float16)vals[c]; h1[c] = (_Float16)vals[c+8]; }
        _Float16* dst = &sF[row*KDIM + g*16];
        *(v8h*)dst       = h0;
        *(v8h*)(dst + 8) = h1;
    }
    __syncthreads();

    // ---- WMMA GEMM ----
    const int  wv   = tid >> 5;
    const int  lane = tid & 31;
    const int  mt   = wv & 3;        // M-tile 0..3
    const int  nh   = wv >> 2;       // N-half 0..1
    const int  lm   = lane & 15;
    const bool lo   = lane < 16;
    const int  arow = mt*16 + lm;

    // Rolled K-loop keeps each accumulator tuple in fixed VGPRs (in-place
    // D=C accumulation, no register-rotation movs / extra hazard NOPs).
    v8f acc0 = {}, acc1 = {}, acc2 = {}, acc3 = {};
    const _Float16* Frow = &sF[arow * KDIM];
    const _Float16* Wrow = Wt + (size_t)(nh*4*16 + lm) * KDIM + (lo ? 0 : 16);
    const int aoff1 = lo ? 0 : 16;
    const int aoff2 = lo ? 8 : 24;

    #pragma unroll 1
    for (int kc = 0; kc < 13; kc++) {
        int kb = kc * 32;
        // A k order used here (contiguous, same as B): lanes<16 hold K 0..15; lanes>=16 hold K 16..31
        v8h a0 = *(const v8h*)(Frow + kb + aoff1);
        v8h a1 = *(const v8h*)(Frow + kb + aoff2);
        v16h av;
        #pragma unroll
        for (int q = 0; q < 8; q++) { av[q] = a0[q]; av[q+8] = a1[q]; }
        // B as loaded here (same k order as A): lane holds column n; lanes<16 K 0..15, lanes>=16 K 16..31
        v16h b0 = *(const v16h*)(Wrow + (size_t)0*16*KDIM + kb);
        v16h b1 = *(const v16h*)(Wrow + (size_t)1*16*KDIM + kb);
        v16h b2 = *(const v16h*)(Wrow + (size_t)2*16*KDIM + kb);
        v16h b3 = *(const v16h*)(Wrow + (size_t)3*16*KDIM + kb);
        acc0 = __builtin_amdgcn_wmma_f32_16x16x32_f16(false, av, false, b0, (short)0, acc0, false, false);
        acc1 = __builtin_amdgcn_wmma_f32_16x16x32_f16(false, av, false, b1, (short)0, acc1, false, false);
        acc2 = __builtin_amdgcn_wmma_f32_16x16x32_f16(false, av, false, b2, (short)0, acc2, false, false);
        acc3 = __builtin_amdgcn_wmma_f32_16x16x32_f16(false, av, false, b3, (short)0, acc3, false, false);
    }
    __syncthreads();   // all waves done reading sF

    // ---- scatter accumulators to LDS, padded stride 129 (bank-conflict-free) ----
    float* Cbuf = (float*)sF;
    {
        int rbase = mt*16 + (lo ? 0 : 8);
        int cbase = nh*4*16 + lm;
        #pragma unroll
        for (int rr = 0; rr < 8; rr++) {
            // C layout: VGPR rr -> row rr (lanes<16) / row rr+8 (lanes>=16), col = lane%16
            Cbuf[(rbase + rr)*129 + cbase + 0*16] = acc0[rr];
            Cbuf[(rbase + rr)*129 + cbase + 1*16] = acc1[rr];
            Cbuf[(rbase + rr)*129 + cbase + 2*16] = acc2[rr];
            Cbuf[(rbase + rr)*129 + cbase + 3*16] = acc3[rr];
        }
    }
    __syncthreads();

    // ---- LayerNorm statistics (one thread per edge row) ----
    if (tid < ROWS) {
        const float* rowp = &Cbuf[tid*129];
        float s = 0.0f;
        for (int c = 0; c < 128; c++) s += rowp[c];
        float mu = s * (1.0f/128.0f);
        float v = 0.0f;
        for (int c = 0; c < 128; c++) { float d = rowp[c]-mu; v += d*d; }
        s_mu[tid] = mu;
        s_rs[tid] = rsqrtf(v*(1.0f/128.0f) + 1e-5f);
    }
    __syncthreads();

    // ---- normalize + coalesced store ----
    for (int idx = tid; idx < ROWS*NDIM; idx += 256) {
        int row = idx >> 7;
        int c   = idx & 127;
        float v = Cbuf[row*129 + c];
        float o = (v - s_mu[row]) * s_rs[row] * ln_g[c] + ln_b[c];
        int gi  = gi0 + row / TOPK;
        int kk  = row % TOPK;
        out[((size_t)gi*TOPK + kk)*NDIM + c] = o;
    }
}

// ---------------------------------------------------------------------------
extern "C" void kernel_launch(void* const* d_in, const int* in_sizes, int n_in,
                              void* d_out, int out_size, void* d_ws, size_t ws_size,
                              hipStream_t stream)
{
    const float* X      = (const float*)d_in[0];
    const float* mask   = (const float*)d_in[1];
    const int*   ridx   = (const int*)d_in[2];
    const int*   chain  = (const int*)d_in[3];
    const float* pe_w   = (const float*)d_in[4];
    const float* pe_b   = (const float*)d_in[5];
    const float* edge_w = (const float*)d_in[6];
    const float* ln_g   = (const float*)d_in[7];
    const float* ln_b   = (const float*)d_in[8];
    (void)n_in; (void)out_size; (void)ws_size;

    const int BL = in_sizes[1];                  // B * L (L fixed at 2048)

    float* outE   = (float*)d_out;
    float* outIdx = outE + (size_t)BL * TOPK * NDIM;   // E_idx as float values

    char* ws = (char*)d_ws;
    int*      ws_eidx = (int*)ws;                                  // BL*30 ints
    float*    ws_dnb  = (float*)(ws + (size_t)BL * TOPK * 4);      // BL*30 floats
    _Float16* ws_Wt   = (_Float16*)(ws + (size_t)2 * BL * TOPK * 4); // 128*416 halves (32B aligned)

    prep_w_kernel<<<(KDIM*NDIM + 255)/256, 256, 0, stream>>>(edge_w, ws_Wt);
    topk_kernel<<<BL, 256, 0, stream>>>(X, mask, ws_eidx, ws_dnb, outIdx);
    feat_kernel<<<BL/RPD, 256, 0, stream>>>(X, ridx, chain, pe_w, pe_b, ln_g, ln_b,
                                            ws_eidx, ws_dnb, ws_Wt, outE);
}
